// CausalSelfAttention_45973329936781
// MI455X (gfx1250) — matrix operand order repaired and checked by reading, not compiled, not run on hardware
//
#include <hip/hip_runtime.h>

// ---------------------------------------------------------------------------
// CDNA5 (gfx1250) causal self-attention: bf16 WMMA GEMMs + flash attention.
// wave32, 8 waves / 256-thread workgroups.
// Double-buffered LDS tiles; row-major tiles streamed with
// global_load_async_to_lds_b128 overlapped under the WMMA chain;
// fragments read as contiguous ds_load_b128 (k = 16h + s, same order in A and B).
// GEMM: 256x64 block tile, 32 rows/wave (2 A-frags x 4 B-frags = 8 WMMA/step).
// ---------------------------------------------------------------------------

typedef __attribute__((ext_vector_type(16))) __bf16 v16bf;
typedef __attribute__((ext_vector_type(8)))  float  v8f;

#define N_HEAD 16
#define N_EMBD 1024
#define HEAD_DIM 64
#define BATCH 4
#define SEQ 2048
#define NTOK (BATCH * SEQ)          // 8192

static __device__ __forceinline__ unsigned short f2bf(float f) {
    unsigned int u = __builtin_bit_cast(unsigned int, f);
    unsigned int r = u + 0x7FFFu + ((u >> 16) & 1u);   // round-to-nearest-even
    return (unsigned short)(r >> 16);
}
static __device__ __forceinline__ unsigned int pack2bf(float lo, float hi) {
    return (unsigned int)f2bf(lo) | ((unsigned int)f2bf(hi) << 16);
}

static __device__ __forceinline__ v8f wmma_bf16(v16bf a, v16bf b, v8f c) {
    return __builtin_amdgcn_wmma_f32_16x16x32_bf16(
        false, a, false, b, (short)0, c, false, false);
}

// Async DMA of one 16B chunk: global (saddr + 32-bit voffset) -> LDS.
static __device__ __forceinline__ void async_g2l_b128(
        unsigned lds_byte_addr, unsigned global_byte_off, const void* sbase) {
    asm volatile("global_load_async_to_lds_b128 %0, %1, %2"
                 :: "v"(lds_byte_addr), "v"(global_byte_off), "s"(sbase)
                 : "memory");
}
static __device__ __forceinline__ void async_wait0() {
    asm volatile("s_wait_asynccnt 0" ::: "memory");
}

// A-fragment (16x32 bf16): lane 0-15: M=lane, kh=0; lane 16-31: M=lane-16,
// kh=16. K = [kh,kh+16): two contiguous 16B chunks.
static __device__ __forceinline__ v16bf lds_load_A16x32(
        const unsigned short* t, int stride, int lane) {
    union { v16bf v; uint4 q[2]; } r;
    const int m  = lane & 15;
    const int kh = (lane < 16) ? 0 : 16;
    const unsigned short* p = t + m * stride + kh;
    r.q[0] = *(const uint4*)(p);
    r.q[1] = *(const uint4*)(p + 8);
    return r.v;
}

// B-fragment (32x16) from a K-contiguous tile Tt[n][k] (transposed storage):
// lane&15 = column n, kh = (lane<16?0:16); 16 contiguous ushorts [kh, kh+16).
static __device__ __forceinline__ v16bf lds_load_Bt32x16(
        const unsigned short* t, int stride, int col0, int lane) {
    union { v16bf v; uint4 q[2]; } r;
    const int n  = col0 + (lane & 15);
    const int kh = (lane < 16) ? 0 : 16;
    const unsigned short* p = t + n * stride + kh;
    r.q[0] = *(const uint4*)(p);
    r.q[1] = *(const uint4*)(p + 8);
    return r.v;
}

// ---------------------------------------------------------------------------
// Kernel 1: fp32 -> bf16 conversion, 8 elements/thread/iter (b128 in/out)
// ---------------------------------------------------------------------------
__global__ __launch_bounds__(256) void cvt_f32_bf16(
        const float4* __restrict__ src, uint4* __restrict__ dst, int n8) {
    for (int g = blockIdx.x * 256 + threadIdx.x; g < n8; g += gridDim.x * 256) {
        float4 a = src[2 * g];
        float4 b = src[2 * g + 1];
        uint4 o;
        o.x = pack2bf(a.x, a.y);
        o.y = pack2bf(a.z, a.w);
        o.z = pack2bf(b.x, b.y);
        o.w = pack2bf(b.z, b.w);
        dst[g] = o;
    }
}

// ---------------------------------------------------------------------------
// Kernel 2/4: tiled bf16 WMMA GEMM  out[8192, N] = A[8192,1024] @ W[1024,N] + b
// Block tile 256x64, 8 waves x 32 rows; K-step 32; double-buffered,
// async-pipelined. Each wave: 2 A-frags x 4 shared B-frags -> 8 WMMA/step.
// mode 0: scatter q/k/v bf16 [B,H,T,64].  mode 1: fp32 out [B,T,C].
// ---------------------------------------------------------------------------
__global__ __launch_bounds__(256) void gemm_bf16(
        const unsigned short* __restrict__ A,
        const unsigned short* __restrict__ W,
        const float* __restrict__ bias,
        int N, int mode,
        unsigned short* __restrict__ qo,
        unsigned short* __restrict__ ko,
        unsigned short* __restrict__ vo,
        float* __restrict__ out) {
    __shared__ alignas(16) unsigned short As[2][256][48];  // row-major [m][k]
    __shared__ alignas(16) unsigned short Bt[2][64][48];   // transposed [n][k]

    const int tid  = threadIdx.x;
    const int wave = tid >> 5;
    const int lane = tid & 31;
    const int m0   = blockIdx.x * 256;
    const int n0   = blockIdx.y * 64;

    const unsigned int* Wu = (const unsigned int*)W;

    const v8f zero8 = {0.f, 0.f, 0.f, 0.f, 0.f, 0.f, 0.f, 0.f};
    v8f acc[2][4];
#pragma unroll
    for (int s = 0; s < 2; ++s)
#pragma unroll
        for (int c = 0; c < 4; ++c) acc[s][c] = zero8;

    // ---- prologue: stage tile kb=0 into buffer 0 ----
#pragma unroll
    for (int p = 0; p < 4; ++p) {
        int i   = tid + p * 256;
        int row = i >> 2;
        int c   = i & 3;
        unsigned lds  = (unsigned)(size_t)&As[0][row][c * 8];
        unsigned goff = ((unsigned)(m0 + row) * (unsigned)N_EMBD) * 2u +
                        (unsigned)c * 16u;
        async_g2l_b128(lds, goff, (const void*)A);
    }
#pragma unroll
    for (int p = 0; p < 4; ++p) {
        int i    = tid + p * 256;
        int krow = i >> 5;
        int cp   = i & 31;
        unsigned int d = Wu[((krow * N + n0) >> 1) + cp];
        Bt[0][2 * cp][krow]     = (unsigned short)(d & 0xffffu);
        Bt[0][2 * cp + 1][krow] = (unsigned short)(d >> 16);
    }

    int cur = 0;
    for (int kb = 0; kb < N_EMBD; kb += 32) {
        async_wait0();         // A-tile DMA for `cur` complete
        __syncthreads();       // all waves' DMA + W ds-stores visible
        const int  nxt  = cur ^ 1;
        const bool more = (kb + 32 < N_EMBD);
        unsigned wreg[4];

        if (more) {
            // issue next A tile DMA (overlaps with compute below)
#pragma unroll
            for (int p = 0; p < 4; ++p) {
                int i   = tid + p * 256;
                int row = i >> 2;
                int c   = i & 3;
                unsigned lds  = (unsigned)(size_t)&As[nxt][row][c * 8];
                unsigned goff = ((unsigned)(m0 + row) * (unsigned)N_EMBD +
                                 (unsigned)(kb + 32)) * 2u + (unsigned)c * 16u;
                async_g2l_b128(lds, goff, (const void*)A);
            }
            // issue next W tile global loads (stores happen after compute)
#pragma unroll
            for (int p = 0; p < 4; ++p) {
                int i    = tid + p * 256;
                int krow = i >> 5;
                int cp   = i & 31;
                wreg[p] = Wu[(((kb + 32 + krow) * N + n0) >> 1) + cp];
            }
            __builtin_prefetch(&A[(m0 + tid) * N_EMBD + kb + 64], 0, 0);
        }

        // ---- compute from `cur`: batch frag loads, then WMMA chain ----
        v16bf a0 = lds_load_A16x32(&As[cur][wave * 32][0],      48, lane);
        v16bf a1 = lds_load_A16x32(&As[cur][wave * 32 + 16][0], 48, lane);
        v16bf bf[4];
#pragma unroll
        for (int c = 0; c < 4; ++c)
            bf[c] = lds_load_Bt32x16(&Bt[cur][0][0], 48, c * 16, lane);
#pragma unroll
        for (int c = 0; c < 4; ++c) {
            acc[0][c] = wmma_bf16(a0, bf[c], acc[0][c]);
            acc[1][c] = wmma_bf16(a1, bf[c], acc[1][c]);
        }

        if (more) {            // scatter staged W regs transposed into nxt
#pragma unroll
            for (int p = 0; p < 4; ++p) {
                int i    = tid + p * 256;
                int krow = i >> 5;
                int cp   = i & 31;
                Bt[nxt][2 * cp][krow]     = (unsigned short)(wreg[p] & 0xffffu);
                Bt[nxt][2 * cp + 1][krow] = (unsigned short)(wreg[p] >> 16);
            }
        }
        cur = nxt;
    }

    // epilogue: C/D layout -> lane&15 = N, M = VGPR j + 8*(lane>=16)
    const int nl = lane & 15;
    const int mh = (lane >> 4) << 3;
#pragma unroll
    for (int s = 0; s < 2; ++s) {
#pragma unroll
        for (int c = 0; c < 4; ++c) {
#pragma unroll
            for (int j = 0; j < 8; ++j) {
                int col = n0 + c * 16 + nl;
                int row = m0 + wave * 32 + s * 16 + j + mh;
                float val = acc[s][c][j] + bias[col];
                if (mode == 0) {
                    int which = col >> 10;          // 0=q 1=k 2=v
                    int cc    = col & 1023;
                    int h     = cc >> 6;
                    int d     = cc & 63;
                    int b     = row >> 11;
                    int t     = row & 2047;
                    unsigned short* dst =
                        (which == 0) ? qo : ((which == 1) ? ko : vo);
                    dst[(((b << 4) + h) * SEQ + t) * HEAD_DIM + d] = f2bf(val);
                } else {
                    out[(size_t)row * N + col] = val;
                }
            }
        }
    }
}

// ---------------------------------------------------------------------------
// Kernel 3: flash attention. grid = (T/128, H, B), block = 256 (8 waves).
// Double-buffered K (async DMA, row-major) and V (staged transposed) tiles.
// ---------------------------------------------------------------------------
__global__ __launch_bounds__(256) void flash_attn(
        const unsigned short* __restrict__ q,   // [B,H,T,64] bf16
        const unsigned short* __restrict__ k,
        const unsigned short* __restrict__ v,
        unsigned short* __restrict__ y) {       // [B,T,C] bf16
    __shared__ alignas(16) unsigned short Ks[2][32][48];     // [key][d]
    __shared__ alignas(16) unsigned short Vt[2][64][48];     // [d][key]
    __shared__ alignas(16) unsigned short Pls[8][16][48];    // per-wave P tile

    const int tid  = threadIdx.x;
    const int wave = tid >> 5;
    const int lane = tid & 31;
    const int qt   = blockIdx.x;
    const int h    = blockIdx.y;
    const int b    = blockIdx.z;
    const int qbase = qt * 128 + wave * 16;

    const unsigned int* vu = (const unsigned int*)v;
    const int headrow = (b * N_HEAD + h) * SEQ;

    // Q fragments (d = 0..31, 32..63), A layout, b128 global loads
    v16bf qf[2];
    {
        const int m  = lane & 15;
        const int kh = (lane < 16) ? 0 : 16;
        const unsigned short* qrow = q + (size_t)(headrow + qbase + m) * HEAD_DIM;
#pragma unroll
        for (int f = 0; f < 2; ++f) {
            union { v16bf v; uint4 qq[2]; } r;
            const unsigned short* p = qrow + f * 32 + kh;
            r.qq[0] = *(const uint4*)(p);
            r.qq[1] = *(const uint4*)(p + 8);
            qf[f] = r.v;
        }
    }

    const v8f zero8 = {0.f, 0.f, 0.f, 0.f, 0.f, 0.f, 0.f, 0.f};
    v8f Oacc[4];
#pragma unroll
    for (int c = 0; c < 4; ++c) Oacc[c] = zero8;
    float mst[8], lst[8];
#pragma unroll
    for (int j = 0; j < 8; ++j) { mst[j] = -3.0e38f; lst[j] = 0.f; }

    const int nl = lane & 15;
    const int mh = (lane >> 4) << 3;
    const int kblocks = (qt + 1) * 4;                  // keys < (qt+1)*128

    // ---- prologue: stage key block 0 into buffer 0 ----
    if (tid < 128) {                                   // waves 0-3: K DMA
        int row = tid >> 2;
        int c   = tid & 3;
        unsigned lds  = (unsigned)(size_t)&Ks[0][row][c * 8];
        unsigned goff = ((unsigned)(headrow + row) * (unsigned)HEAD_DIM +
                         (unsigned)c * 8u) * 2u;
        async_g2l_b128(lds, goff, (const void*)k);
    }
#pragma unroll
    for (int p = 0; p < 4; ++p) {
        int i   = tid + p * 256;
        int key = i >> 5;
        int cp  = i & 31;
        unsigned int d = vu[(size_t)(headrow + key) * 32 + cp];
        Vt[0][2 * cp][key]     = (unsigned short)(d & 0xffffu);
        Vt[0][2 * cp + 1][key] = (unsigned short)(d >> 16);
    }

    int cur = 0;
    for (int kb = 0; kb < kblocks; ++kb) {
        const int key0 = kb * 32;
        async_wait0();
        __syncthreads();
        const int  nxt  = cur ^ 1;
        const bool more = (kb + 1 < kblocks);
        unsigned vreg[4];

        if (more) {
            if (tid < 128) {
                int row = tid >> 2;
                int c   = tid & 3;
                unsigned lds  = (unsigned)(size_t)&Ks[nxt][row][c * 8];
                unsigned goff = ((unsigned)(headrow + key0 + 32 + row) *
                                 (unsigned)HEAD_DIM + (unsigned)c * 8u) * 2u;
                async_g2l_b128(lds, goff, (const void*)k);
            }
#pragma unroll
            for (int p = 0; p < 4; ++p) {
                int i   = tid + p * 256;
                int key = i >> 5;
                int cp  = i & 31;
                vreg[p] = vu[(size_t)(headrow + key0 + 32 + key) * 32 + cp];
            }
        }

        if (key0 <= qbase + 15) {                      // wave-uniform
            // S[16q x 32keys] = Q . K^T : two 16x16 tiles, contiguous K-frags
            v8f s[2];
#pragma unroll
            for (int kc = 0; kc < 2; ++kc) {
                const int key = kc * 16 + nl;
                const int dh  = (lane < 16) ? 0 : 16;
                union { v16bf v; uint4 qq[2]; } b0, b1;
                const unsigned short* kr = &Ks[cur][key][0];
                b0.qq[0] = *(const uint4*)(kr + dh);
                b0.qq[1] = *(const uint4*)(kr + dh + 8);
                b1.qq[0] = *(const uint4*)(kr + 32 + dh);
                b1.qq[1] = *(const uint4*)(kr + 32 + dh + 8);
                s[kc] = wmma_bf16(qf[1], b1.v, wmma_bf16(qf[0], b0.v, zero8));
            }

            // causal mask + online softmax (fp32), 16-lane row reductions
            float p0[8], p1[8];
#pragma unroll
            for (int j = 0; j < 8; ++j) {
                const int qrow = qbase + j + mh;
                float s0 = s[0][j] * 0.125f;           // 1/sqrt(64)
                float s1 = s[1][j] * 0.125f;
                if (key0 + nl      > qrow) s0 = -1.0e30f;
                if (key0 + 16 + nl > qrow) s1 = -1.0e30f;
                float rm = fmaxf(s0, s1);
#pragma unroll
                for (int off = 8; off >= 1; off >>= 1)
                    rm = fmaxf(rm, __shfl_xor(rm, off, 16));
                const float mnew  = fmaxf(mst[j], rm);
                const float alpha = __expf(mst[j] - mnew);
                const float e0 = __expf(s0 - mnew);
                const float e1 = __expf(s1 - mnew);
                float rs = e0 + e1;
#pragma unroll
                for (int off = 8; off >= 1; off >>= 1)
                    rs += __shfl_xor(rs, off, 16);
                lst[j] = lst[j] * alpha + rs;
                mst[j] = mnew;
#pragma unroll
                for (int c = 0; c < 4; ++c) Oacc[c][j] *= alpha;
                p0[j] = e0;
                p1[j] = e1;
            }

            // bounce P through LDS to re-lay as A fragment (16x32)
#pragma unroll
            for (int j = 0; j < 8; ++j) {
                const int pr = j + mh;
                Pls[wave][pr][nl]      = f2bf(p0[j]);
                Pls[wave][pr][16 + nl] = f2bf(p1[j]);
            }
            v16bf pf = lds_load_A16x32(&Pls[wave][0][0], 48, lane);

            // O (16x64) += P (16x32) . V (32x64), contiguous V-frags
            v16bf vb[4];
#pragma unroll
            for (int c = 0; c < 4; ++c)
                vb[c] = lds_load_Bt32x16(&Vt[cur][0][0], 48, c * 16, lane);
#pragma unroll
            for (int c = 0; c < 4; ++c)
                Oacc[c] = wmma_bf16(pf, vb[c], Oacc[c]);
        }

        if (more) {            // scatter staged V regs transposed into nxt
#pragma unroll
            for (int p = 0; p < 4; ++p) {
                int i   = tid + p * 256;
                int key = i >> 5;
                int cp  = i & 31;
                Vt[nxt][2 * cp][key]     = (unsigned short)(vreg[p] & 0xffffu);
                Vt[nxt][2 * cp + 1][key] = (unsigned short)(vreg[p] >> 16);
            }
        }
        cur = nxt;
    }

    // normalize and write y [B,T,C] as bf16
#pragma unroll
    for (int c = 0; c < 4; ++c) {
#pragma unroll
        for (int j = 0; j < 8; ++j) {
            const int t = qbase + j + mh;
            float val = Oacc[c][j] / lst[j];
            y[((size_t)(b * SEQ + t)) * N_EMBD + h * HEAD_DIM + c * 16 + nl] =
                f2bf(val);
        }
    }
}

// ---------------------------------------------------------------------------
// launch
// ---------------------------------------------------------------------------
extern "C" void kernel_launch(void* const* d_in, const int* in_sizes, int n_in,
                              void* d_out, int out_size, void* d_ws, size_t ws_size,
                              hipStream_t stream) {
    const float* x      = (const float*)d_in[0];  // [4,2048,1024]
    const float* W_attn = (const float*)d_in[1];  // [1024,3072]
    const float* b_attn = (const float*)d_in[2];  // [3072]
    const float* W_proj = (const float*)d_in[3];  // [1024,1024]
    const float* b_proj = (const float*)d_in[4];  // [1024]
    float* out = (float*)d_out;                   // [4,2048,1024]

    char* ws = (char*)d_ws;
    const size_t n_x  = (size_t)NTOK * N_EMBD;
    const size_t n_wa = (size_t)N_EMBD * 3 * N_EMBD;
    const size_t n_wp = (size_t)N_EMBD * N_EMBD;
    const size_t n_hd = (size_t)BATCH * N_HEAD * SEQ * HEAD_DIM;

    unsigned short* xb  = (unsigned short*)(ws);
    unsigned short* wab = (unsigned short*)(ws + 2 * n_x);
    unsigned short* wpb = (unsigned short*)(ws + 2 * (n_x + n_wa));
    unsigned short* qb  = (unsigned short*)(ws + 2 * (n_x + n_wa + n_wp));
    unsigned short* kb  = qb + n_hd;
    unsigned short* vb  = kb + n_hd;
    unsigned short* yb  = vb + n_hd;

    cvt_f32_bf16<<<1024, 256, 0, stream>>>((const float4*)x,      (uint4*)xb,
                                           (int)(n_x  / 8));
    cvt_f32_bf16<<<1024, 256, 0, stream>>>((const float4*)W_attn, (uint4*)wab,
                                           (int)(n_wa / 8));
    cvt_f32_bf16<<<256,  256, 0, stream>>>((const float4*)W_proj, (uint4*)wpb,
                                           (int)(n_wp / 8));

    gemm_bf16<<<dim3(NTOK / 256, (3 * N_EMBD) / 64), 256, 0, stream>>>(
        xb, wab, b_attn, 3 * N_EMBD, 0, qb, kb, vb, nullptr);

    flash_attn<<<dim3(SEQ / 128, N_HEAD, BATCH), 256, 0, stream>>>(qb, kb, vb, yb);

    gemm_bf16<<<dim3(NTOK / 256, N_EMBD / 64), 256, 0, stream>>>(
        yb, wpb, b_proj, N_EMBD, 1, nullptr, nullptr, nullptr, out);
}
